// InterCrossAttention_S_84524956385359
// MI455X (gfx1250) — compile-verified, matrix operand order checked by reading, not run on hardware
//
#include <hip/hip_runtime.h>
#include <hip/hip_bf16.h>

// Shapes from the reference
// R=256 rows (l), C=256 cols (j/i), DZ=64, H=4, DHC=8, DC=32.
// Wbias/bbias are dropped: the bias is constant over the softmax axis,
// and softmax is shift-invariant, so it cancels exactly.

typedef __attribute__((ext_vector_type(16))) _Float16 v16h;
typedef __attribute__((ext_vector_type(8)))  _Float16 h8;
typedef __attribute__((ext_vector_type(8)))  float    v8f;
typedef int i4v __attribute__((vector_size(4 * sizeof(int))));

union V16 { v16h v; h8 h[2]; };

__device__ __forceinline__ v8f wmma_f32_f16(v16h a, v16h b, v8f c) {
  // D = A(16x32 f16) x B(32x16 f16) + C(16x16 f32)
  return __builtin_amdgcn_wmma_f32_16x16x32_f16(false, a, false, b,
                                                (short)0, c, false, false);
}

// ---------------------------------------------------------------------------
// Kernel 1: LayerNorm + Q/K/V/Gate projections (WMMA), write f16 intermediates
// ---------------------------------------------------------------------------
__global__ __launch_bounds__(256) void icsa_proj(
    const float* __restrict__ zi, const float* __restrict__ za,
    const float* __restrict__ lng, const float* __restrict__ lnb,
    const float* __restrict__ Wq, const float* __restrict__ bq,
    const float* __restrict__ Wk, const float* __restrict__ bk,
    const float* __restrict__ Wv, const float* __restrict__ bv,
    const float* __restrict__ Wg, const float* __restrict__ bg,
    _Float16* __restrict__ q_ws, _Float16* __restrict__ k_ws,
    _Float16* __restrict__ vT_ws, _Float16* __restrict__ g_ws)
{
  __shared__ _Float16 wT[4][32][64];         // [mat][out][in] transposed weights
  __shared__ float    wb[4][32];             // biases
  __shared__ _Float16 atile[8][2][16][64];   // [wave][zi/za][token][chan]

  const int tid  = threadIdx.x;
  const int wave = tid >> 5;
  const int lane = tid & 31;

  for (int idx = tid; idx < 4 * 64 * 32; idx += 256) {
    const int w    = idx >> 11;
    const int rem  = idx & 2047;
    const int cin  = rem >> 5;
    const int cout = rem & 31;
    const float* W = (w == 0) ? Wq : (w == 1) ? Wk : (w == 2) ? Wv : Wg;
    wT[w][cout][cin] = (_Float16)W[cin * 32 + cout];
  }
  if (tid < 128) {
    const int w = tid >> 5, c = tid & 31;
    const float* B = (w == 0) ? bq : (w == 1) ? bk : (w == 2) ? bv : bg;
    wb[w][c] = B[c];
  }
  __syncthreads();

  const int t0   = (blockIdx.x * 8 + wave) * 16;  // 16-token tile
  const int l    = t0 >> 8;
  const int j0   = t0 & 255;
  const int mrow = lane & 15;
  const int half = (lane >> 4) & 1;
  const int klo  = half * 8;
  const int ch0  = half * 32;

  // LayerNorm: lane pair (L, L+16) covers the 64 channels of token mrow
  #pragma unroll
  for (int src = 0; src < 2; ++src) {
    const float* z = (src == 0) ? zi : za;
    const float* p = z + (size_t)(t0 + mrow) * 64 + ch0;
    float x[32];
    float s = 0.f;
    #pragma unroll
    for (int c = 0; c < 32; ++c) { x[c] = p[c]; s += x[c]; }
    s += __shfl_xor(s, 16, 32);
    const float mu = s * (1.f / 64.f);
    float vv = 0.f;
    #pragma unroll
    for (int c = 0; c < 32; ++c) { const float d = x[c] - mu; vv += d * d; }
    vv += __shfl_xor(vv, 16, 32);
    const float rstd = rsqrtf(vv * (1.f / 64.f) + 1e-5f);
    #pragma unroll
    for (int c = 0; c < 32; ++c) {
      atile[wave][src][mrow][ch0 + c] =
          (_Float16)((x[c] - mu) * rstd * lng[ch0 + c] + lnb[ch0 + c]);
    }
  }
  asm volatile("s_wait_dscnt 0" ::: "memory");  // own-wave LDS produce->consume

  const float qscale = 0.35355339059327373f;  // 1/sqrt(8)
  #pragma unroll
  for (int w = 0; w < 4; ++w) {
    const int src = (w == 1 || w == 2) ? 1 : 0;  // k,v from za; q,g from zi
    #pragma unroll
    for (int n = 0; n < 2; ++n) {
      v8f acc;
      #pragma unroll
      for (int e = 0; e < 8; ++e) acc[e] = 0.f;
      #pragma unroll
      for (int s = 0; s < 2; ++s) {
        V16 a, b;
        a.h[0] = *(const h8*)&atile[wave][src][mrow][32 * s + klo];
        a.h[1] = *(const h8*)&atile[wave][src][mrow][32 * s + klo + 16];
        const int col = 16 * n + mrow;
        b.h[0] = *(const h8*)&wT[w][col][32 * s + klo];
        b.h[1] = *(const h8*)&wT[w][col][32 * s + klo + 16];
        acc = wmma_f32_f16(a.v, b.v, acc);
      }
      const int chan = 16 * n + mrow;   // output channel (lane-resident)
      const int hh   = chan >> 3;
      const int cc   = chan & 7;
      const float bias = wb[w][chan];
      #pragma unroll
      for (int r = 0; r < 8; ++r) {
        const int row = r + half * 8;   // token within tile
        const int tok = t0 + row;
        const float dvl = acc[r] + bias;
        if (w == 0) {
          q_ws[(((size_t)(l * 4 + hh)) * 256 + (j0 + row)) * 8 + cc] =
              (_Float16)(dvl * qscale);
        } else if (w == 1) {
          k_ws[(((size_t)(l * 4 + hh)) * 256 + (j0 + row)) * 8 + cc] =
              (_Float16)dvl;
        } else if (w == 2) {
          vT_ws[(((size_t)(l * 4 + hh)) * 8 + cc) * 256 + (j0 + row)] =
              (_Float16)dvl;
        } else {
          g_ws[(size_t)tok * 32 + chan] =
              (_Float16)(1.f / (1.f + __expf(-dvl)));
        }
      }
    }
  }
}

// ---------------------------------------------------------------------------
// Kernel 2: flash attention per (l, h, 16-query tile); one wave per tile.
// All 8 waves of a block share (l, h): stage K and V^T tiles to LDS once per
// block via CDNA5 async-to-LDS (ASYNCcnt) and read B-matrices from LDS.
// ---------------------------------------------------------------------------
__global__ __launch_bounds__(256) void icsa_attn(
    const _Float16* __restrict__ q_ws, const _Float16* __restrict__ k_ws,
    const _Float16* __restrict__ vT_ws, const _Float16* __restrict__ g_ws,
    _Float16* __restrict__ o_ws)
{
  __shared__ _Float16 sp[8][16][32];   // per-wave P tile (j x i-chunk)
  __shared__ _Float16 sK[256 * 8];     // K   [i][c]  (4 KB)
  __shared__ _Float16 sV[8 * 256];     // V^T [c][i]  (4 KB)

  const int tid  = threadIdx.x;
  const int wave = tid >> 5;
  const int lane = tid & 31;
  const int gw   = blockIdx.x * 8 + wave;   // 0..16383
  const int jt   = gw & 15;
  const int h    = (gw >> 4) & 3;
  const int l    = gw >> 6;
  const int mrow = lane & 15;
  const int half = (lane >> 4) & 1;
  const int klo  = half * 8;
  const int j0   = jt * 16;

  // (l,h) is block-uniform: lh = blockIdx.x >> 1
  const int lh = blockIdx.x >> 1;
  const _Float16* kg = k_ws + (size_t)lh * 2048;   // 256*8 halfs
  const _Float16* vg = vT_ws + (size_t)lh * 2048;  // 8*256 halfs

#if __has_builtin(__builtin_amdgcn_global_load_async_to_lds_b128)
  {
    // 256 lanes x 16B = 4KB per instruction; one for K, one for V^T
    __builtin_amdgcn_global_load_async_to_lds_b128(
        (i4v*)(kg + tid * 8), (i4v*)(sK + tid * 8), 0, 0);
    __builtin_amdgcn_global_load_async_to_lds_b128(
        (i4v*)(vg + tid * 8), (i4v*)(sV + tid * 8), 0, 0);
    asm volatile("s_wait_asynccnt 0" ::: "memory");
  }
#else
  {
    *(h8*)(sK + tid * 8) = *(const h8*)(kg + tid * 8);
    *(h8*)(sV + tid * 8) = *(const h8*)(vg + tid * 8);
  }
#endif
  __syncthreads();

  const _Float16* qbase = q_ws + ((size_t)(l * 4 + h)) * 256 * 8;

  // Q as A-matrix, head dim (8) zero-padded to K=32 via masked loads
  V16 qa;
  #pragma unroll
  for (int e = 0; e < 16; ++e) qa.v[e] = (_Float16)0.f;
  if (half == 0) qa.h[0] = *(const h8*)(qbase + (size_t)(j0 + mrow) * 8);

  // Padded B-matrix registers: zero lanes/halves are loop-invariant
  V16 bk0, bk1, bv_;
  #pragma unroll
  for (int e = 0; e < 16; ++e) {
    bk0.v[e] = (_Float16)0.f;
    bk1.v[e] = (_Float16)0.f;
    bv_.v[e] = (_Float16)0.f;
  }

  float mrun[8], lrun[8];
  v8f oacc;
  #pragma unroll
  for (int r = 0; r < 8; ++r) { mrun[r] = -1e30f; lrun[r] = 0.f; oacc[r] = 0.f; }

  for (int it = 0; it < 8; ++it) {
    const int i0 = it * 32;
    if (half == 0) {
      bk0.h[0] = *(const h8*)&sK[(i0 + mrow) * 8];
      bk1.h[0] = *(const h8*)&sK[(i0 + 16 + mrow) * 8];
    }
    v8f zf;
    #pragma unroll
    for (int e = 0; e < 8; ++e) zf[e] = 0.f;
    v8f s0 = wmma_f32_f16(qa.v, bk0.v, zf);   // scores, i in [i0, i0+16)
    v8f s1 = wmma_f32_f16(qa.v, bk1.v, zf);   // scores, i in [i0+16, i0+32)

    // streaming softmax; row M = r + 8*half lives in one 16-lane half
    #pragma unroll
    for (int r = 0; r < 8; ++r) {
      float t = fmaxf(s0[r], s1[r]);
      t = fmaxf(t, __shfl_xor(t, 1, 32));
      t = fmaxf(t, __shfl_xor(t, 2, 32));
      t = fmaxf(t, __shfl_xor(t, 4, 32));
      t = fmaxf(t, __shfl_xor(t, 8, 32));
      const float mn    = fmaxf(mrun[r], t);
      const float alpha = __expf(mrun[r] - mn);
      mrun[r] = mn;
      const float p0 = __expf(s0[r] - mn);
      const float p1 = __expf(s1[r] - mn);
      float rs = p0 + p1;
      rs += __shfl_xor(rs, 1, 32);
      rs += __shfl_xor(rs, 2, 32);
      rs += __shfl_xor(rs, 4, 32);
      rs += __shfl_xor(rs, 8, 32);
      lrun[r] = lrun[r] * alpha + rs;
      oacc[r] *= alpha;
      const int M = r + half * 8;
      sp[wave][M][mrow]      = (_Float16)p0;   // D-layout -> LDS
      sp[wave][M][16 + mrow] = (_Float16)p1;
    }
    asm volatile("s_wait_dscnt 0" ::: "memory");

    V16 pa;  // P reloaded in A-layout (16 x 32)
    pa.h[0] = *(const h8*)&sp[wave][mrow][klo];
    pa.h[1] = *(const h8*)&sp[wave][mrow][klo + 16];

    if (mrow < 8) {  // V chunk as B-matrix: N = head channel (8 real + pad)
      bv_.h[0] = *(const h8*)&sV[mrow * 256 + i0 + klo];
      bv_.h[1] = *(const h8*)&sV[mrow * 256 + i0 + klo + 16];
    }
    oacc = wmma_f32_f16(pa.v, bv_.v, oacc);
    asm volatile("" ::: "memory");
  }

  if (mrow < 8) {   // lane column = head channel; apply 1/l, gate, store gated
    #pragma unroll
    for (int r = 0; r < 8; ++r) {
      const int j = j0 + r + half * 8;
      const float inv = 1.f / lrun[r];
      const size_t gi = ((size_t)(l * 256 + j)) * 32 + h * 8 + mrow;
      const float gv = (float)g_ws[gi];
      o_ws[gi] = (_Float16)(oacc[r] * inv * gv);
    }
  }
}

// ---------------------------------------------------------------------------
// Kernel 3: gated output @ Wo + bo  ([65536 x 32] x [32 x 64])
// ---------------------------------------------------------------------------
__global__ __launch_bounds__(256) void icsa_oproj(
    const _Float16* __restrict__ o_ws, const float* __restrict__ Wo,
    const float* __restrict__ bo, float* __restrict__ out)
{
  __shared__ _Float16 wT[64][32];  // [out][in]
  __shared__ float bb[64];
  const int tid  = threadIdx.x;
  const int wave = tid >> 5;
  const int lane = tid & 31;
  for (int idx = tid; idx < 64 * 32; idx += 256) {
    const int cout = idx & 63, cin = idx >> 6;
    wT[cout][cin] = (_Float16)Wo[cin * 64 + cout];
  }
  if (tid < 64) bb[tid] = bo[tid];
  __syncthreads();

  const int t0   = (blockIdx.x * 8 + wave) * 16;
  const int mrow = lane & 15;
  const int half = (lane >> 4) & 1;
  const int klo  = half * 8;

  V16 a;
  a.h[0] = *(const h8*)(o_ws + (size_t)(t0 + mrow) * 32 + klo);
  a.h[1] = *(const h8*)(o_ws + (size_t)(t0 + mrow) * 32 + klo + 16);

  #pragma unroll
  for (int n = 0; n < 4; ++n) {
    const int col = 16 * n + mrow;
    V16 b;
    b.h[0] = *(const h8*)&wT[col][klo];
    b.h[1] = *(const h8*)&wT[col][klo + 16];
    v8f acc;
    #pragma unroll
    for (int e = 0; e < 8; ++e) acc[e] = 0.f;
    acc = wmma_f32_f16(a.v, b.v, acc);
    const float bias = bb[col];
    #pragma unroll
    for (int r = 0; r < 8; ++r) {
      const int tok = t0 + r + half * 8;
      out[(size_t)tok * 64 + col] = acc[r] + bias;
    }
  }
}

// ---------------------------------------------------------------------------
extern "C" void kernel_launch(void* const* d_in, const int* in_sizes, int n_in,
                              void* d_out, int out_size, void* d_ws, size_t ws_size,
                              hipStream_t stream) {
  (void)in_sizes; (void)n_in; (void)out_size; (void)ws_size;
  const float* zi  = (const float*)d_in[0];
  const float* za  = (const float*)d_in[1];
  const float* lng = (const float*)d_in[2];
  const float* lnb = (const float*)d_in[3];
  const float* Wq  = (const float*)d_in[4];
  const float* bq  = (const float*)d_in[5];
  const float* Wk  = (const float*)d_in[6];
  const float* bk  = (const float*)d_in[7];
  const float* Wv  = (const float*)d_in[8];
  const float* bv  = (const float*)d_in[9];
  // d_in[10]=Wbias, d_in[11]=bbias: per-query constant over softmax axis ->
  // mathematically cancels (softmax shift invariance); intentionally unused.
  const float* Wg  = (const float*)d_in[12];
  const float* bg  = (const float*)d_in[13];
  const float* Wo  = (const float*)d_in[14];
  const float* bo  = (const float*)d_in[15];
  float* out = (float*)d_out;

  char* ws = (char*)d_ws;
  const size_t SEG = (size_t)4 * 1024 * 1024;
  _Float16* q_ws  = (_Float16*)(ws + 0 * SEG);  // [l][h][j][8]
  _Float16* k_ws  = (_Float16*)(ws + 1 * SEG);  // [l][h][i][8]
  _Float16* vT_ws = (_Float16*)(ws + 2 * SEG);  // [l][h][c][i]
  _Float16* g_ws  = (_Float16*)(ws + 3 * SEG);  // [tok][32] sigmoid(gate)
  _Float16* o_ws  = (_Float16*)(ws + 4 * SEG);  // [tok][32] gated attn out

  icsa_proj<<<512, 256, 0, stream>>>(zi, za, lng, lnb, Wq, bq, Wk, bk, Wv, bv,
                                     Wg, bg, q_ws, k_ws, vT_ws, g_ws);
  icsa_attn<<<2048, 256, 0, stream>>>(q_ws, k_ws, vT_ws, g_ws, o_ws);
  icsa_oproj<<<512, 256, 0, stream>>>(o_ws, Wo, bo, out);
}
